// GAT_35734127903472
// MI455X (gfx1250) — matrix operand order repaired and checked by reading, not compiled, not run on hardware
//
#include <hip/hip_runtime.h>
#include <hip/hip_bf16.h>

#define H 256
#define NSEQ 4096
#define BATCH 4

typedef __attribute__((ext_vector_type(16))) _Float16 v16h;
typedef __attribute__((ext_vector_type(8)))  _Float16 v8h;
typedef __attribute__((ext_vector_type(4)))  _Float16 v4h;
typedef __attribute__((ext_vector_type(8)))  float    v8f;
typedef __attribute__((ext_vector_type(4)))  float    v4f;

static __device__ inline v8f wmma16(v16h a, v16h b, v8f c) {
  return __builtin_amdgcn_wmma_f32_16x16x32_f16(false, a, false, b, (short)0, c, false, false);
}

static __device__ inline void split4(v4f q, int base, v16h& hi, v16h& lo) {
#pragma unroll
  for (int i = 0; i < 4; ++i) {
    float v = q[i];
    _Float16 h = (_Float16)v;
    hi[base + i] = h;
    lo[base + i] = (_Float16)(v - (float)h);
  }
}

// Build A-fragment (hi/lo split) from 16 f32 values: rows m=lane%16,
// k = i + (lane>=16?16:0). p points at the row base; g = lane>>4.
static __device__ inline void build_a_from_f32(const float* p, int k0, int g,
                                               v16h& a_hi, v16h& a_lo) {
  v4f q0 = *(const v4f*)(p + k0 + g * 16);
  v4f q1 = *(const v4f*)(p + k0 + g * 16 + 4);
  v4f q2 = *(const v4f*)(p + k0 + g * 16 + 8);
  v4f q3 = *(const v4f*)(p + k0 + g * 16 + 12);
  split4(q0, 0, a_hi, a_lo);
  split4(q1, 4, a_hi, a_lo);
  split4(q2, 8, a_hi, a_lo);
  split4(q3, 12, a_hi, a_lo);
}

// ---------------- kernel 1: fold Wc, split Wv/Wmlp into f16 hi/lo ----------------
__global__ void k_prep(const float* __restrict__ Wa, const float* __restrict__ ba,
                       const float* __restrict__ Wb, const float* __restrict__ bb,
                       const float* __restrict__ Wv, const float* __restrict__ Wmlp,
                       const float* __restrict__ Wc, const float* __restrict__ bc,
                       float* ua, float* ub, float* consts,
                       _Float16* Wv_hi, _Float16* Wv_lo,
                       _Float16* Wm_hi, _Float16* Wm_lo) {
  int g = blockIdx.x * 256 + threadIdx.x;   // 0..65535
  {
    float w = Wv[g];
    _Float16 h = (_Float16)w;
    Wv_hi[g] = h;
    Wv_lo[g] = (_Float16)(w - (float)h);
    float w2 = Wmlp[g];
    _Float16 h2 = (_Float16)w2;
    Wm_hi[g] = h2;
    Wm_lo[g] = (_Float16)(w2 - (float)h2);
  }
  if (g < H) {
    float da = 0.f, db = 0.f;
    for (int k = 0; k < H; ++k) {
      da += Wa[k * H + g] * Wc[k];
      db += Wb[k * H + g] * Wc[H + k];
    }
    ua[g] = da;
    ub[g] = db;
  }
  if (g == 0) {
    float ca = 0.f, cb = 0.f;
    for (int k = 0; k < H; ++k) {
      ca += ba[k] * Wc[k];
      cb += bb[k] * Wc[H + k];
    }
    consts[0] = ca + bc[0];   // folded into sa
    consts[1] = cb;
  }
}

// ---------------- kernel 2: sa/sb row dots (one wave per row) ----------------
__global__ void k_scores(const float* __restrict__ x, const float* __restrict__ ua,
                         const float* __restrict__ ub, const float* __restrict__ consts,
                         float* sa, float* sb) {
  __shared__ float uas[H];
  __shared__ float ubs[H];
  int t = threadIdx.x;
  uas[t] = ua[t];
  ubs[t] = ub[t];
  __syncthreads();
  int wave = t >> 5, lane = t & 31;
  int row = blockIdx.x * 8 + wave;          // < BATCH*NSEQ
  const float* xr = x + (size_t)row * H;
  float da = 0.f, db = 0.f;
#pragma unroll
  for (int i = 0; i < 8; ++i) {
    int idx = lane + i * 32;
    float xv = xr[idx];
    da += xv * uas[idx];
    db += xv * ubs[idx];
  }
#pragma unroll
  for (int m = 16; m >= 1; m >>= 1) {
    da += __shfl_xor(da, m, 32);
    db += __shfl_xor(db, m, 32);
  }
  if (lane == 0) {
    sa[row] = da + consts[0];
    sb[row] = db + consts[1];
  }
}

// ---------------- kernel 2b: per-batch max of sb ----------------
__global__ void k_sbmax(const float* __restrict__ sb, float* sbmax) {
  __shared__ float red[256];
  int b = blockIdx.x, t = threadIdx.x;
  const float* p = sb + (size_t)b * NSEQ;
  float m = -3.0e38f;
  for (int c = t; c < NSEQ; c += 256) m = fmaxf(m, p[c]);
  red[t] = m;
  __syncthreads();
  for (int s = 128; s >= 1; s >>= 1) {
    if (t < s) red[t] = fmaxf(red[t], red[t + s]);
    __syncthreads();
  }
  if (t == 0) sbmax[b] = red[0];
}

// ---------------- kernel 3: v = x@Wv.T + bv, stored transposed f16 hi/lo ----------------
__global__ void k_v(const float* __restrict__ x, const _Float16* __restrict__ Wv_hi,
                    const _Float16* __restrict__ Wv_lo, const float* __restrict__ bv,
                    _Float16* Vt_hi, _Float16* Vt_lo) {
  int c0 = blockIdx.x * 16;                 // flattened row (b*NSEQ + c), tile of 16
  int b = c0 >> 12;
  int cin = c0 & (NSEQ - 1);
  int t = threadIdx.x;
  int wave = t >> 5, lane = t & 31;
  int n_ = lane & 15, g = lane >> 4;
  int h0 = wave * 32;                       // this wave: h-tiles h0 and h0+16
  v8f acc0 = {}, acc1 = {};
  const float* xrow = x + (size_t)(c0 + n_) * H;
  for (int k0 = 0; k0 < H; k0 += 32) {
    v16h a_hi, a_lo;
    build_a_from_f32(xrow, k0, g, a_hi, a_lo);
    const _Float16* b0h = Wv_hi + (size_t)(h0 + n_) * H + k0 + g * 16;
    const _Float16* b0l = Wv_lo + (size_t)(h0 + n_) * H + k0 + g * 16;
    v16h bh0 = *(const v16h*)b0h;
    v16h bl0 = *(const v16h*)b0l;
    v16h bh1 = *(const v16h*)(b0h + 16 * H);
    v16h bl1 = *(const v16h*)(b0l + 16 * H);
    acc0 = wmma16(a_hi, bh0, acc0);
    acc0 = wmma16(a_lo, bh0, acc0);
    acc0 = wmma16(a_hi, bl0, acc0);
    acc1 = wmma16(a_hi, bh1, acc1);
    acc1 = wmma16(a_lo, bh1, acc1);
    acc1 = wmma16(a_hi, bl1, acc1);
  }
#pragma unroll
  for (int j = 0; j < 2; ++j) {
    v8f acc = j ? acc1 : acc0;
    int hh = h0 + j * 16 + n_;
    float bvn = bv[hh];
    v8h sh, sl;
#pragma unroll
    for (int r = 0; r < 8; ++r) {
      float v = acc[r] + bvn;
      _Float16 hv = (_Float16)v;
      sh[r] = hv;
      sl[r] = (_Float16)(v - (float)hv);
    }
    size_t o = (size_t)(b * H + hh) * NSEQ + cin + g * 8;
    *(v8h*)(Vt_hi + o) = sh;
    *(v8h*)(Vt_lo + o) = sl;
  }
}

// ---------------- kernel 4: single-pass softmax(attn) @ V via WMMA ----------------
// 64 rows (4 l-tiles) per workgroup for 4x V-reuse; double-buffered P in LDS.
__global__ void k_attn(const float* __restrict__ sa, const float* __restrict__ sb,
                       const float* __restrict__ sbmax,
                       const _Float16* __restrict__ Vt_hi,
                       const _Float16* __restrict__ Vt_lo, float* O) {
  __shared__ _Float16 Phi[2][4][16][72];
  __shared__ _Float16 Plo[2][4][16][72];
  __shared__ float zpart[64][5];
  __shared__ float zinv[64];
  int bt = blockIdx.x;
  int b = bt >> 6;                  // NSEQ/64 = 64 tiles per batch
  int l0 = (bt & 63) << 6;
  int t = threadIdx.x;
  // phase-A role: thread owns row rr (0..63) and k-slot of 16 columns
  int rr = t & 63;
  int alt = rr >> 4, am = rr & 15;
  int ks = t >> 6;                  // 0..3
  float A_r = sa[b * NSEQ + l0 + rr];            // includes ca + bc
  float tmx = A_r + sbmax[b];
  float mrow = tmx >= 0.f ? tmx : 0.01f * tmx;   // lrelu monotone => exact row max
  const float* sbb = sb + (size_t)b * NSEQ;
  float zacc = 0.f;
  // phase-B role
  int wave = t >> 5, lane = t & 31;
  int n_ = lane & 15, g = lane >> 4;
  int h0 = wave * 32;
  const v8f vzero = {};
  v8f acc[4][2];
#pragma unroll
  for (int lt = 0; lt < 4; ++lt) {
    acc[lt][0] = vzero;
    acc[lt][1] = vzero;
  }
  const _Float16* v0h = Vt_hi + (size_t)(b * H + h0 + n_) * NSEQ;
  const _Float16* v0l = Vt_lo + (size_t)(b * H + h0 + n_) * NSEQ;

  for (int c0 = 0; c0 < NSEQ; c0 += 64) {
    int buf = (c0 >> 6) & 1;
    // ---- phase A: P = exp(lrelu(A_r + sb_c) - mrow), pre-split hi/lo ----
    {
      _Float16* ph = &Phi[buf][alt][am][ks * 16];
      _Float16* pl = &Plo[buf][alt][am][ks * 16];
#pragma unroll
      for (int q = 0; q < 2; ++q) {
        v4f s0 = *(const v4f*)(sbb + c0 + ks * 16 + q * 8);
        v4f s1 = *(const v4f*)(sbb + c0 + ks * 16 + q * 8 + 4);
        v8h hv, lv;
#pragma unroll
        for (int j2 = 0; j2 < 4; ++j2) {
          float tt = A_r + s0[j2];
          tt = tt >= 0.f ? tt : 0.01f * tt;
          float p = __expf(tt - mrow);
          zacc += p;
          _Float16 hh = (_Float16)p;
          hv[j2] = hh;
          lv[j2] = (_Float16)(p - (float)hh);
          float tt2 = A_r + s1[j2];
          tt2 = tt2 >= 0.f ? tt2 : 0.01f * tt2;
          float p2 = __expf(tt2 - mrow);
          zacc += p2;
          _Float16 hh2 = (_Float16)p2;
          hv[4 + j2] = hh2;
          lv[4 + j2] = (_Float16)(p2 - (float)hh2);
        }
        *(v8h*)(ph + q * 8) = hv;
        *(v8h*)(pl + q * 8) = lv;
      }
    }
    __syncthreads();
    // ---- phase B: 2 K-subchunks of 32, 2 h-tiles x 4 l-tiles, 3 WMMAs each ----
#pragma unroll
    for (int s = 0; s < 2; ++s) {
      int coff = c0 + s * 32 + g * 16;
      v16h b0h = *(const v16h*)(v0h + coff);
      v16h b0l = *(const v16h*)(v0l + coff);
      v16h b1h = *(const v16h*)(v0h + 16 * NSEQ + coff);
      v16h b1l = *(const v16h*)(v0l + 16 * NSEQ + coff);
#pragma unroll
      for (int lt = 0; lt < 4; ++lt) {
        v16h a_hi, a_lo;
        const _Float16* p0 = &Phi[buf][lt][n_][s * 32 + g * 16];
        const _Float16* q0 = &Plo[buf][lt][n_][s * 32 + g * 16];
        v8h x0 = *(const v8h*)p0;
        v8h x1 = *(const v8h*)(p0 + 8);
        v8h y0 = *(const v8h*)q0;
        v8h y1 = *(const v8h*)(q0 + 8);
#pragma unroll
        for (int i = 0; i < 8; ++i) {
          a_hi[i] = x0[i]; a_hi[8 + i] = x1[i];
          a_lo[i] = y0[i]; a_lo[8 + i] = y1[i];
        }
        acc[lt][0] = wmma16(a_hi, b0h, acc[lt][0]);
        acc[lt][0] = wmma16(a_lo, b0h, acc[lt][0]);
        acc[lt][0] = wmma16(a_hi, b0l, acc[lt][0]);
        acc[lt][1] = wmma16(a_hi, b1h, acc[lt][1]);
        acc[lt][1] = wmma16(a_lo, b1h, acc[lt][1]);
        acc[lt][1] = wmma16(a_hi, b1l, acc[lt][1]);
      }
    }
    // no trailing barrier: next phase A writes the other buffer; the next
    // iteration's barrier orders reuse of this buffer two chunks later.
  }
  __syncthreads();
  // ---- reduce Z across k-slots, then normalize + write numerator ----
  zpart[rr][ks] = zacc;
  __syncthreads();
  if (t < 64) {
    zinv[t] = 1.0f / (zpart[t][0] + zpart[t][1] + zpart[t][2] + zpart[t][3]);
  }
  __syncthreads();
#pragma unroll
  for (int lt = 0; lt < 4; ++lt) {
#pragma unroll
    for (int j = 0; j < 2; ++j) {
      int hh = h0 + j * 16 + n_;
#pragma unroll
      for (int r = 0; r < 8; ++r) {
        int mm = r + g * 8;
        int row = l0 + lt * 16 + mm;
        O[((size_t)(b * NSEQ + row)) * H + hh] = acc[lt][j][r] * zinv[lt * 16 + mm];
      }
    }
  }
}

// ---------------- kernel 5: out = tanh(O@Wmlp.T + bmlp) + x ----------------
__global__ void k_mlp(const float* __restrict__ O, const _Float16* __restrict__ Wm_hi,
                      const _Float16* __restrict__ Wm_lo, const float* __restrict__ bmlp,
                      const float* __restrict__ x, float* out) {
  int r0 = blockIdx.x * 16;                 // flattened row tile
  int t = threadIdx.x;
  int wave = t >> 5, lane = t & 31;
  int n_ = lane & 15, g = lane >> 4;
  int d0 = wave * 32;
  v8f acc0 = {}, acc1 = {};
  const float* orow = O + (size_t)(r0 + n_) * H;
  for (int k0 = 0; k0 < H; k0 += 32) {
    v16h a_hi, a_lo;
    build_a_from_f32(orow, k0, g, a_hi, a_lo);
    const _Float16* b0h = Wm_hi + (size_t)(d0 + n_) * H + k0 + g * 16;
    const _Float16* b0l = Wm_lo + (size_t)(d0 + n_) * H + k0 + g * 16;
    v16h bh0 = *(const v16h*)b0h;
    v16h bl0 = *(const v16h*)b0l;
    v16h bh1 = *(const v16h*)(b0h + 16 * H);
    v16h bl1 = *(const v16h*)(b0l + 16 * H);
    acc0 = wmma16(a_hi, bh0, acc0);
    acc0 = wmma16(a_lo, bh0, acc0);
    acc0 = wmma16(a_hi, bl0, acc0);
    acc1 = wmma16(a_hi, bh1, acc1);
    acc1 = wmma16(a_lo, bh1, acc1);
    acc1 = wmma16(a_hi, bl1, acc1);
  }
#pragma unroll
  for (int j = 0; j < 2; ++j) {
    v8f acc = j ? acc1 : acc0;
    int dd = d0 + j * 16 + n_;
    float bn = bmlp[dd];
#pragma unroll
    for (int r = 0; r < 8; ++r) {
      int mm = r + g * 8;
      size_t idx = (size_t)(r0 + mm) * H + dd;
      out[idx] = tanhf(acc[r] + bn) + x[idx];
    }
  }
}

extern "C" void kernel_launch(void* const* d_in, const int* in_sizes, int n_in,
                              void* d_out, int out_size, void* d_ws, size_t ws_size,
                              hipStream_t stream) {
  (void)in_sizes; (void)n_in; (void)out_size; (void)ws_size;
  const float* x    = (const float*)d_in[0];
  const float* Wa   = (const float*)d_in[1];
  const float* ba   = (const float*)d_in[2];
  const float* Wb   = (const float*)d_in[3];
  const float* bb   = (const float*)d_in[4];
  const float* Wv   = (const float*)d_in[5];
  const float* bv   = (const float*)d_in[6];
  const float* Wc   = (const float*)d_in[7];
  const float* bc   = (const float*)d_in[8];
  const float* Wmlp = (const float*)d_in[9];
  const float* bmlp = (const float*)d_in[10];
  float* out = (float*)d_out;

  char* ws = (char*)d_ws;
  size_t off = 0;
  auto alloc = [&](size_t bytes) {
    size_t o = off;
    off += (bytes + 255) & ~(size_t)255;
    return o;
  };
  float*    ua     = (float*)(ws + alloc(H * 4));
  float*    ub     = (float*)(ws + alloc(H * 4));
  float*    consts = (float*)(ws + alloc(256));
  _Float16* Wv_hi  = (_Float16*)(ws + alloc((size_t)H * H * 2));
  _Float16* Wv_lo  = (_Float16*)(ws + alloc((size_t)H * H * 2));
  _Float16* Wm_hi  = (_Float16*)(ws + alloc((size_t)H * H * 2));
  _Float16* Wm_lo  = (_Float16*)(ws + alloc((size_t)H * H * 2));
  float*    sa     = (float*)(ws + alloc((size_t)BATCH * NSEQ * 4));
  float*    sb     = (float*)(ws + alloc((size_t)BATCH * NSEQ * 4));
  float*    sbm    = (float*)(ws + alloc(256));
  _Float16* Vt_hi  = (_Float16*)(ws + alloc((size_t)BATCH * H * NSEQ * 2));
  _Float16* Vt_lo  = (_Float16*)(ws + alloc((size_t)BATCH * H * NSEQ * 2));
  float*    O      = (float*)(ws + alloc((size_t)BATCH * NSEQ * H * 4));

  k_prep<<<(H * H) / 256, 256, 0, stream>>>(Wa, ba, Wb, bb, Wv, Wmlp, Wc, bc,
                                            ua, ub, consts, Wv_hi, Wv_lo, Wm_hi, Wm_lo);
  k_scores<<<(BATCH * NSEQ) / 8, 256, 0, stream>>>(x, ua, ub, consts, sa, sb);
  k_sbmax<<<BATCH, 256, 0, stream>>>(sb, sbm);
  k_v<<<(BATCH * NSEQ) / 16, 256, 0, stream>>>(x, Wv_hi, Wv_lo, bv, Vt_hi, Vt_lo);
  k_attn<<<BATCH * (NSEQ / 64), 256, 0, stream>>>(sa, sb, sbm, Vt_hi, Vt_lo, O);
  k_mlp<<<(BATCH * NSEQ) / 16, 256, 0, stream>>>(O, Wm_hi, Wm_lo, bmlp, x, out);
}
